// FeedForward_45621142618737
// MI455X (gfx1250) — matrix operand order repaired and checked by reading, not compiled, not run on hardware
//
#include <hip/hip_runtime.h>
#include <hip/hip_bf16.h>

// ---------------------------------------------------------------------------
// NF4 SwiGLU FFN for gfx1250 (MI455X): fused NF4->bf16 dequant + WMMA GEMMs
// with double-buffered LDS staging and async global->LDS DMA (ASYNCcnt).
//   k_gate_up : h = silu(x @ W1^T) * (x @ W3^T)   (bf16 h staged in d_ws)
//   k_down    : out = h @ W2^T                     (f32 out)
// ---------------------------------------------------------------------------

typedef __attribute__((ext_vector_type(16))) __bf16 v16bf;
typedef __attribute__((ext_vector_type(8)))  __bf16 v8bf;
typedef __attribute__((ext_vector_type(8)))  float  v8f;
typedef __attribute__((ext_vector_type(4)))  int    v4i;

union V16 { v16bf v; v8bf h[2]; };

constexpr int DIM = 4096;    // model dim
constexpr int HID = 11008;   // hidden dim
constexpr int MT  = 128;     // workgroup M tile
constexpr int NT  = 64;      // workgroup N tile
constexpr int KT  = 64;      // K tile == NF4 block size (one absmax / row / tile)
constexpr int KP  = 72;      // padded A row stride (elements) -> 144B, 16B aligned
constexpr int NP  = 72;      // padded B^T row stride (elements)

#if __has_builtin(__builtin_amdgcn_global_load_async_to_lds_b128) && \
    __has_builtin(__builtin_amdgcn_s_wait_asynccnt)
#define USE_ASYNC_LDS 1
#else
#define USE_ASYNC_LDS 0
#endif

#if USE_ASYNC_LDS
// 16B global -> LDS DMA, tracked by ASYNCcnt (no VGPR data staging).
__device__ __forceinline__ void async_cp16(const void* gsrc, void* ldst) {
    __builtin_amdgcn_global_load_async_to_lds_b128(
        (__attribute__((address_space(1))) v4i*)gsrc,
        (__attribute__((address_space(3))) v4i*)ldst, 0, 0);
}
#endif

__constant__ float NF4_TAB[16] = {
    -1.0f, -0.6961928009986877f, -0.5250730514526367f, -0.39491748809814453f,
    -0.28444138169288635f, -0.18477343022823334f, -0.09105003625154495f, 0.0f,
    0.07958029955625534f, 0.16093020141124725f, 0.24611230194568634f,
    0.33791524171829224f, 0.44070982933044434f, 0.5626170039176941f,
    0.6989939212799072f, 1.0f };

// WMMA bf16 fragment helpers (wave32 layouts per CDNA5 ISA 7.12.2)
// A 16x32: lane m = lane&15; lanes<16 hold K {0..7,16..23}, lanes>=16 {8..15,24..31}
// B 32x16 (stored K-major in LDS): lane n = lane&15; same K order as A (strided gather).
__device__ __forceinline__ V16 load_a_frag(const __bf16* As, int mrow, int ks, int lhi) {
    V16 a;
    const __bf16* p = &As[mrow * KP + ks + lhi];
    a.h[0] = *(const v8bf*)(p);
    a.h[1] = *(const v8bf*)(p + 16);
    return a;
}
__device__ __forceinline__ V16 load_b_frag(const __bf16* Bt, int krow, int ncol) {
    V16 b;
    const __bf16* p = &Bt[krow * NP + ncol];
    #pragma unroll
    for (int s = 0; s < 8; ++s) {
        b.v[s]     = p[s * NP];
        b.v[8 + s] = p[(16 + s) * NP];
    }
    return b;
}

// ---------------------------------------------------------------------------
// Kernel 1: h[m, n] = silu(sum_k x[m,k] W1[n,k]) * (sum_k x[m,k] W3[n,k])
// ---------------------------------------------------------------------------
__global__ __launch_bounds__(256)
void k_gate_up(const float* __restrict__ x,
               const int*  __restrict__ w1c, const float* __restrict__ w1a,
               const int*  __restrict__ w3c, const float* __restrict__ w3a,
               __bf16* __restrict__ hout)
{
    __shared__ alignas(16) __bf16 As[2][MT * KP];  // x tile, bf16, [m][k]
    __shared__ alignas(16) __bf16 B1[2][KT * NP];  // W1 tile, K-major [k][n]
    __shared__ alignas(16) __bf16 B3[2][KT * NP];  // W3 tile, K-major [k][n]
    __shared__ float nf4s[16];

    const int tid = threadIdx.x;
    const int m0  = blockIdx.y * MT;
    const int n0  = blockIdx.x * NT;
    if (tid < 16) nf4s[tid] = NF4_TAB[tid];

    const int wave = tid >> 5, lane = tid & 31;
    const int wm   = (wave & 3) * 32;     // 4 waves along M
    const int wn   = (wave >> 2) * 32;    // 2 waves along N
    const int lmod = lane & 15;
    const int lhi  = (lane >> 4) * 8;

    v8f acc1[2][2] = {};
    v8f acc3[2][2] = {};

    auto stage = [&](int k0, int buf) {
        // ---- A: 128x64 f32 -> bf16 (2048 float4 loads, 8/thread)
        #pragma unroll
        for (int i = 0; i < 8; ++i) {
            int g   = tid + i * 256;
            int row = g >> 4;
            int cq  = (g & 15) * 4;
            const float4 f = *(const float4*)(x + (size_t)(m0 + row) * DIM + k0 + cq);
            __bf16* dst = &As[buf][row * KP + cq];
            dst[0] = (__bf16)f.x; dst[1] = (__bf16)f.y;
            dst[2] = (__bf16)f.z; dst[3] = (__bf16)f.w;
        }
        // ---- B1/B3: dequant 64x64 codes, store transposed (K-major)
        #pragma unroll
        for (int i = 0; i < 4; ++i) {
            int g  = tid + i * 256;
            int n  = g >> 4;
            int kq = (g & 15) * 4;
            size_t cidx = (size_t)(n0 + n) * DIM + k0 + kq;
            int    aidx = (n0 + n) * (DIM / 64) + (k0 >> 6);
            const int4 c1 = *(const int4*)(w1c + cidx);
            const float a1 = w1a[aidx];
            B1[buf][(kq + 0) * NP + n] = (__bf16)(nf4s[c1.x & 15] * a1);
            B1[buf][(kq + 1) * NP + n] = (__bf16)(nf4s[c1.y & 15] * a1);
            B1[buf][(kq + 2) * NP + n] = (__bf16)(nf4s[c1.z & 15] * a1);
            B1[buf][(kq + 3) * NP + n] = (__bf16)(nf4s[c1.w & 15] * a1);
            const int4 c3 = *(const int4*)(w3c + cidx);
            const float a3 = w3a[aidx];
            B3[buf][(kq + 0) * NP + n] = (__bf16)(nf4s[c3.x & 15] * a3);
            B3[buf][(kq + 1) * NP + n] = (__bf16)(nf4s[c3.y & 15] * a3);
            B3[buf][(kq + 2) * NP + n] = (__bf16)(nf4s[c3.z & 15] * a3);
            B3[buf][(kq + 3) * NP + n] = (__bf16)(nf4s[c3.w & 15] * a3);
            if (k0 + KT < DIM) {
                __builtin_prefetch(w1c + cidx + KT, 0, 1);
                __builtin_prefetch(w3c + cidx + KT, 0, 1);
            }
        }
    };

    __syncthreads();           // publish nf4s
    stage(0, 0);

    const int T = DIM / KT;
    for (int t = 0; t < T; ++t) {
        __syncthreads();       // buf[t&1] staged; buf[(t+1)&1] free
        if (t + 1 < T) stage((t + 1) * KT, (t + 1) & 1);
        const int buf = t & 1;
        #pragma unroll
        for (int ks = 0; ks < KT; ks += 32) {
            V16 af[2];
            #pragma unroll
            for (int mt = 0; mt < 2; ++mt)
                af[mt] = load_a_frag(As[buf], wm + mt * 16 + lmod, ks, lhi);
            V16 b1f[2], b3f[2];
            #pragma unroll
            for (int nt = 0; nt < 2; ++nt) {
                b1f[nt] = load_b_frag(B1[buf], ks + lhi, wn + nt * 16 + lmod);
                b3f[nt] = load_b_frag(B3[buf], ks + lhi, wn + nt * 16 + lmod);
            }
            #pragma unroll
            for (int mt = 0; mt < 2; ++mt)
                #pragma unroll
                for (int nt = 0; nt < 2; ++nt) {
                    acc1[mt][nt] = __builtin_amdgcn_wmma_f32_16x16x32_bf16(
                        false, af[mt].v, false, b1f[nt].v, (short)0, acc1[mt][nt], false, false);
                    acc3[mt][nt] = __builtin_amdgcn_wmma_f32_16x16x32_bf16(
                        false, af[mt].v, false, b3f[nt].v, (short)0, acc3[mt][nt], false, false);
                }
        }
    }

    // ---- epilogue: silu(gate) * up -> bf16 h
    #pragma unroll
    for (int mt = 0; mt < 2; ++mt)
        #pragma unroll
        for (int nt = 0; nt < 2; ++nt) {
            const int col = n0 + wn + nt * 16 + lmod;
            #pragma unroll
            for (int r = 0; r < 8; ++r) {
                const int m = m0 + wm + mt * 16 + lhi + r;
                const float g = acc1[mt][nt][r];
                const float s = g / (1.0f + __expf(-g));
                hout[(size_t)m * HID + col] = (__bf16)(s * acc3[mt][nt][r]);
            }
        }
}

// ---------------------------------------------------------------------------
// Kernel 2: out[m, d] = sum_h h[m, h] * W2[d, h]
// ---------------------------------------------------------------------------
__global__ __launch_bounds__(256)
void k_down(const __bf16* __restrict__ hmat,
            const int*   __restrict__ w2c, const float* __restrict__ w2a,
            float* __restrict__ out)
{
    __shared__ alignas(16) __bf16 As[2][MT * KP];
    __shared__ alignas(16) __bf16 Bs[2][KT * NP];
    __shared__ float nf4s[16];

    const int tid = threadIdx.x;
    const int m0  = blockIdx.y * MT;
    const int n0  = blockIdx.x * NT;
    if (tid < 16) nf4s[tid] = NF4_TAB[tid];

    const int wave = tid >> 5, lane = tid & 31;
    const int wm   = (wave & 3) * 32;
    const int wn   = (wave >> 2) * 32;
    const int lmod = lane & 15;
    const int lhi  = (lane >> 4) * 8;

    v8f acc[2][2] = {};

    auto stage = [&](int k0, int buf) {
        // ---- A: 128x64 bf16 copy; async DMA to LDS when the builtin exists
        #pragma unroll
        for (int i = 0; i < 4; ++i) {
            int g   = tid + i * 256;
            int row = g >> 3;
            int cq  = (g & 7) * 8;
            const __bf16* src = hmat + (size_t)(m0 + row) * HID + k0 + cq;
            __bf16* dst = &As[buf][row * KP + cq];
#if USE_ASYNC_LDS
            async_cp16(src, dst);
#else
            *(uint4*)dst = *(const uint4*)src;
            if (k0 + KT < HID) __builtin_prefetch(src + KT, 0, 1);
#endif
        }
        // ---- B: dequant W2 64x64 codes, K-major
        #pragma unroll
        for (int i = 0; i < 4; ++i) {
            int g  = tid + i * 256;
            int n  = g >> 4;
            int kq = (g & 15) * 4;
            size_t cidx = (size_t)(n0 + n) * HID + k0 + kq;
            int    aidx = (n0 + n) * (HID / 64) + (k0 >> 6);
            const int4 c = *(const int4*)(w2c + cidx);
            const float a = w2a[aidx];
            Bs[buf][(kq + 0) * NP + n] = (__bf16)(nf4s[c.x & 15] * a);
            Bs[buf][(kq + 1) * NP + n] = (__bf16)(nf4s[c.y & 15] * a);
            Bs[buf][(kq + 2) * NP + n] = (__bf16)(nf4s[c.z & 15] * a);
            Bs[buf][(kq + 3) * NP + n] = (__bf16)(nf4s[c.w & 15] * a);
            if (k0 + KT < HID) __builtin_prefetch(w2c + cidx + KT, 0, 1);
        }
    };

    __syncthreads();           // publish nf4s
    stage(0, 0);
#if USE_ASYNC_LDS
    __builtin_amdgcn_s_wait_asynccnt(0);
#endif

    const int T = HID / KT;
    for (int t = 0; t < T; ++t) {
        __syncthreads();       // buf[t&1] staged; buf[(t+1)&1] free
        if (t + 1 < T) stage((t + 1) * KT, (t + 1) & 1);   // DMA overlaps compute below
        const int buf = t & 1;
        #pragma unroll
        for (int ks = 0; ks < KT; ks += 32) {
            V16 af[2];
            #pragma unroll
            for (int mt = 0; mt < 2; ++mt)
                af[mt] = load_a_frag(As[buf], wm + mt * 16 + lmod, ks, lhi);
            V16 bf[2];
            #pragma unroll
            for (int nt = 0; nt < 2; ++nt)
                bf[nt] = load_b_frag(Bs[buf], ks + lhi, wn + nt * 16 + lmod);
            #pragma unroll
            for (int mt = 0; mt < 2; ++mt)
                #pragma unroll
                for (int nt = 0; nt < 2; ++nt)
                    acc[mt][nt] = __builtin_amdgcn_wmma_f32_16x16x32_bf16(
                        false, af[mt].v, false, bf[nt].v, (short)0, acc[mt][nt], false, false);
        }
#if USE_ASYNC_LDS
        __builtin_amdgcn_s_wait_asynccnt(0);   // my DMA for t+1 done before next barrier
#endif
    }

    #pragma unroll
    for (int mt = 0; mt < 2; ++mt)
        #pragma unroll
        for (int nt = 0; nt < 2; ++nt) {
            const int col = n0 + wn + nt * 16 + lmod;
            #pragma unroll
            for (int r = 0; r < 8; ++r) {
                const int m = m0 + wm + mt * 16 + lhi + r;
                out[(size_t)m * DIM + col] = acc[mt][nt][r];
            }
        }
}

// ---------------------------------------------------------------------------
extern "C" void kernel_launch(void* const* d_in, const int* in_sizes, int n_in,
                              void* d_out, int out_size, void* d_ws, size_t ws_size,
                              hipStream_t stream)
{
    const float* x   = (const float*)d_in[0];
    const int*   w1c = (const int*)  d_in[1];
    const float* w1a = (const float*)d_in[2];
    const int*   w2c = (const int*)  d_in[3];
    const float* w2a = (const float*)d_in[4];
    const int*   w3c = (const int*)  d_in[5];
    const float* w3a = (const float*)d_in[6];

    const int M = in_sizes[0] / DIM;          // B*S tokens (4096)
    __bf16* hws = (__bf16*)d_ws;              // [M, HID] bf16 intermediate (~90 MB)

    dim3 blk(256);
    k_gate_up<<<dim3(HID / NT, M / MT), blk, 0, stream>>>(x, w1c, w1a, w3c, w3a, hws);
    k_down  <<<dim3(DIM / NT, M / MT), blk, 0, stream>>>(hws, w2c, w2a, (float*)d_out);
}
